// HyperGATModule_54185307406982
// MI455X (gfx1250) — matrix operand order repaired and checked by reading, not compiled, not run on hardware
//
#include <hip/hip_runtime.h>
#include <math.h>

// ---------------- problem constants (match reference setup_inputs) ----------
constexpr int Bc = 4;
constexpr int Nn = 8000;
constexpr int Rr = 16;
constexpr int Ee = 1024;
constexpr int Hh = 128;
constexpr int Ll = 2;
constexpr int Tt = Nn + Rr;        // 8016 (= 501*16)
constexpr int Tp = 8064;           // T padded to multiple of 64 (K dim)
constexpr int KC = 1344;           // LDS K-chunk for hyper GEMM (8064 = 6*1344)
constexpr int NCHUNK = Tp / KC;    // 6
constexpr float SCALE = 0.08838834764831845f;  // 1/sqrt(128)
constexpr float NEGV  = -1e9f;
constexpr float EPSV  = 1e-5f;

// ---------------- WMMA types / helpers --------------------------------------
typedef __bf16 v16bf __attribute__((ext_vector_type(16)));
typedef __bf16 v8bf  __attribute__((ext_vector_type(8)));
typedef float  v8f   __attribute__((ext_vector_type(8)));
typedef unsigned int v4u __attribute__((ext_vector_type(4)));
typedef int v4i __attribute__((ext_vector_type(4)));
typedef int v8i __attribute__((ext_vector_type(8)));

#if defined(__has_builtin)
#if __has_builtin(__builtin_amdgcn_tensor_load_to_lds) && \
    __has_builtin(__builtin_amdgcn_s_wait_tensorcnt)
#define USE_TDM 1
#endif
#endif

static __device__ inline v8f wmma_bf16(v16bf a, v16bf b, v8f c) {
  // D = A(16x32) * B(32x16) + C, fp32 accumulate
  return __builtin_amdgcn_wmma_f32_16x16x32_bf16(false, a, false, b, (short)0, c,
                                                 false, false);
}

static __device__ inline v8f zero8() {
  v8f z = {0.f, 0.f, 0.f, 0.f, 0.f, 0.f, 0.f, 0.f};
  return z;
}

// A-fragment: lane L holds row M=L%16; halves j=0..15 -> K = k0+(L/16)*16 + j.
// Two contiguous 16B chunks.
// `p` must already include row base + k0 + (lane>>4)*16.
static __device__ inline v16bf load_a_frag(const __bf16* p) {
  v8bf lo = *(const v8bf*)(p);
  v8bf hi = *(const v8bf*)(p + 8);
  v16bf a;
#pragma unroll
  for (int i = 0; i < 8; ++i) { a[i] = lo[i]; a[i + 8] = hi[i]; }
  return a;
}

// ---------------- small prep kernels ---------------------------------------
__global__ __launch_bounds__(128)
void k_region(const float* __restrict__ node, const int* __restrict__ mem,
              const float* __restrict__ rinit, float* __restrict__ hA) {
  int r = blockIdx.x, h = threadIdx.x;
  float sum = 0.f; int cnt = 0;
  for (int n = 0; n < Nn; ++n) {
    if (mem[n] == r) { sum += node[(size_t)n * Hh + h]; ++cnt; }
  }
  float reg = (cnt > 0) ? sum / (float)cnt : 0.f;
  hA[(size_t)(Nn + r) * Hh + h] = reg + rinit[(size_t)r * Hh + h];
}

__global__ void k_copy_f32(const float* __restrict__ in, float* __restrict__ out, int n) {
  for (int i = blockIdx.x * blockDim.x + threadIdx.x; i < n; i += gridDim.x * blockDim.x)
    out[i] = in[i];
}

__global__ void k_cvt_bf16(const float* __restrict__ in, __bf16* __restrict__ out, int n) {
  int i = blockIdx.x * blockDim.x + threadIdx.x;
  if (i < n) out[i] = (__bf16)in[i];
}

__global__ __launch_bounds__(128)
void k_c1(const float* __restrict__ W1hat, const float* __restrict__ ctx,
          float* __restrict__ c1) {
  int h = threadIdx.x;
  float acc = 0.f;
  for (int k = 0; k < Hh; ++k) acc += W1hat[(size_t)k * Hh + h] * ctx[k];
  c1[h] = acc;
}

// raw[t] = scale * <hA[t], c1>; global max; u[t] = exp(raw - gmax)
__global__ __launch_bounds__(1024)
void k_raw_u(const float* __restrict__ hA, const float* __restrict__ c1,
             float* __restrict__ u) {
  __shared__ float sc1[Hh];
  __shared__ float red[1024];
  int tid = threadIdx.x;
  if (tid < Hh) sc1[tid] = c1[tid];
  __syncthreads();
  float lm = -INFINITY;
  for (int t = tid; t < Tt; t += 1024) {
    const float* row = hA + (size_t)t * Hh;
    float acc = 0.f;
    for (int h = 0; h < Hh; ++h) acc += row[h] * sc1[h];
    acc *= SCALE;
    u[t] = acc;
    lm = fmaxf(lm, acc);
  }
  red[tid] = lm;
  __syncthreads();
  for (int s = 512; s > 0; s >>= 1) {
    if (tid < s) red[tid] = fmaxf(red[tid], red[tid + s]);
    __syncthreads();
  }
  float gmax = red[0];
  for (int t = tid; t < Tt; t += 1024) u[t] = __expf(u[t] - gmax);
}

// AT[e, t] = bf16( H_inc[t,e] * u[t] ), padded with zeros for t in [T, Tp)
__global__ __launch_bounds__(256)
void k_AbfT(const float* __restrict__ Hinc, const float* __restrict__ u,
            __bf16* __restrict__ AT) {
  __shared__ float tile[32][33];
  int tx = threadIdx.x, ty = threadIdx.y;           // block (32,8)
  int t0 = blockIdx.x * 32, e0 = blockIdx.y * 32;
#pragma unroll
  for (int j = 0; j < 4; ++j) {
    int t = t0 + ty + j * 8;
    float v = 0.f;
    if (t < Tt) v = Hinc[(size_t)t * Ee + e0 + tx] * u[t];
    tile[ty + j * 8][tx] = v;
  }
  __syncthreads();
#pragma unroll
  for (int j = 0; j < 4; ++j) {
    int e = e0 + ty + j * 8;
    AT[(size_t)e * Tp + t0 + tx] = (__bf16)tile[tx][ty + j * 8];
  }
}

__global__ __launch_bounds__(256)
void k_denom(const float* __restrict__ Hinc, const float* __restrict__ u,
             float* __restrict__ s) {
  int e = blockIdx.x * 256 + threadIdx.x;
  float d = 0.f;
  for (int t = 0; t < Tt; ++t) d += Hinc[(size_t)t * Ee + e] * u[t];
  s[e] = (d > 0.f) ? 1.f / d : 0.f;
}

__global__ void k_zero_tail(__bf16* __restrict__ p, int ld, int rows, int c0, int c1v) {
  int w = c1v - c0;
  int i = blockIdx.x * blockDim.x + threadIdx.x;
  if (i < rows * w) {
    int r = i / w, c = c0 + i % w;
    p[(size_t)r * ld + c] = (__bf16)0.f;
  }
}

// ---------------- Y = X @ W^T, WMMA bf16 (K = N-out = 128) ------------------
// X: (M,128) bf16 row-major; W: (128,128) bf16 row-major (lane n = W row n = B col n)
__global__ __launch_bounds__(256)
void k_gemm_xwT(const __bf16* __restrict__ X, const __bf16* __restrict__ W,
                __bf16* __restrict__ out_rm, __bf16* __restrict__ out_tr, int ld_tr) {
  int lane = threadIdx.x & 31;
  int wave = threadIdx.x >> 5;            // 8 waves -> 8 n-tiles
  int half = lane >> 4, ln16 = lane & 15;
  int m0 = blockIdx.x * 16;
  int n0 = wave * 16;
  const __bf16* arow = X + (size_t)(m0 + ln16) * Hh + half * 16;
  const __bf16* brow = W + (size_t)(n0 + ln16) * Hh + half * 16;
  v8f c = zero8();
#pragma unroll
  for (int k = 0; k < 4; ++k) {
    v16bf a = load_a_frag(arow + k * 32);
    v16bf b = *(const v16bf*)(brow + k * 32);
    c = wmma_bf16(a, b, c);
  }
#pragma unroll
  for (int r = 0; r < 8; ++r) {
    int m = m0 + r + half * 8;
    int n = n0 + ln16;
    __bf16 v = (__bf16)c[r];
    if (out_rm) out_rm[(size_t)m * Hh + n] = v;
    if (out_tr) out_tr[(size_t)n * ld_tr + m] = v;
  }
}

// ---------------- hyper[e,h] = s[e] * sum_t AT[e,t]*P1T[h,t]  (K = Tp) ------
// A-tile (16 rows of AT, shared by all 8 waves) is staged in LDS via the
// Tensor Data Mover; B-tiles (P1T, 2MB total) stream from global/L2.
__global__ __launch_bounds__(256)
void k_hyper(const __bf16* __restrict__ AT, const __bf16* __restrict__ P1T,
             const float* __restrict__ s, __bf16* __restrict__ hyperbf) {
  int lane = threadIdx.x & 31;
  int wave = threadIdx.x >> 5;
  int half = lane >> 4, ln16 = lane & 15;
  int e0 = blockIdx.x * 16;
  int n0 = wave * 16;                      // h-tile
  const __bf16* brow = P1T + (size_t)(n0 + ln16) * Tp + half * 16;
  v8f c0 = zero8(), c1 = zero8();

#ifdef USE_TDM
  __shared__ __attribute__((aligned(16))) __bf16 ldsA[16 * KC];
  unsigned ldsoff = (unsigned)(unsigned long long)(uintptr_t)(void*)&ldsA[0];
  unsigned long long gbase =
      (unsigned long long)(uintptr_t)AT + ((unsigned long long)e0 * Tp) * 2ull;
  const __bf16* arow_l = &ldsA[(size_t)ln16 * KC + half * 16];
  for (int cidx = 0; cidx < NCHUNK; ++cidx) {
    int k0 = cidx * KC;
    __syncthreads();
    if (wave == 0) {
      unsigned long long ga = gbase + (unsigned long long)k0 * 2ull;
      // D# group 0: count=1, lds_addr, 57-bit global addr, type=2
      v4u g0;
      g0[0] = 1u;
      g0[1] = ldsoff;
      g0[2] = (unsigned)(ga & 0xFFFFFFFFull);
      g0[3] = (unsigned)((ga >> 32) & 0x01FFFFFFull) | 0x80000000u;
      // D# group 1: data_size=2B, tensor 8064x1024 (stride 8064), tile 1344x16
      v8i g1;
      g1[0] = 0x00010000;                              // data_size = 1 (2 bytes)
      g1[1] = (int)(((unsigned)Tp & 0xFFFFu) << 16);   // tensor_dim0 lo16
      g1[2] = (int)(((unsigned)Ee & 0xFFFFu) << 16);   // dim0 hi16=0 | dim1 lo16
      g1[3] = (int)(((unsigned)KC & 0xFFFFu) << 16);   // dim1 hi16=0 | tile_dim0
      g1[4] = 16;                                      // tile_dim1=16, tile_dim2=0
      g1[5] = Tp;                                      // tensor_dim0_stride lo32
      g1[6] = 0;
      g1[7] = 0;
      v4i gz = {0, 0, 0, 0};
#if __clang_major__ >= 23
      v8i gz8 = {0, 0, 0, 0, 0, 0, 0, 0};
      __builtin_amdgcn_tensor_load_to_lds(g0, g1, gz, gz, gz8, 0);
#else
      __builtin_amdgcn_tensor_load_to_lds(g0, g1, gz, gz, 0);
#endif
      __builtin_amdgcn_s_wait_tensorcnt(0);
    }
    __syncthreads();
    for (int kc = 0; kc < KC; kc += 64) {
      v16bf a0 = load_a_frag(arow_l + kc);
      v16bf b0 = *(const v16bf*)(brow + k0 + kc);
      c0 = wmma_bf16(a0, b0, c0);
      v16bf a1 = load_a_frag(arow_l + kc + 32);
      v16bf b1 = *(const v16bf*)(brow + k0 + kc + 32);
      c1 = wmma_bf16(a1, b1, c1);
    }
  }
#else
  const __bf16* arow = AT + (size_t)(e0 + ln16) * Tp + half * 16;
  for (int k0 = 0; k0 < Tp; k0 += 64) {
    __builtin_prefetch((const void*)(arow + k0 + 512), 0, 1);
    __builtin_prefetch((const void*)(brow + k0 + 512), 0, 1);
    v16bf a0 = load_a_frag(arow + k0);
    v16bf b0 = *(const v16bf*)(brow + k0);
    c0 = wmma_bf16(a0, b0, c0);
    v16bf a1 = load_a_frag(arow + k0 + 32);
    v16bf b1 = *(const v16bf*)(brow + k0 + 32);
    c1 = wmma_bf16(a1, b1, c1);
  }
#endif

#pragma unroll
  for (int r = 0; r < 8; ++r) {
    int e = e0 + r + half * 8;
    float v = (c0[r] + c1[r]) * s[e];
    hyperbf[(size_t)e * Hh + n0 + ln16] = (__bf16)v;
  }
}

// ---------------- fused stage 2: score -> softmax*Hinc -> @hv -> LN --------
// P3:(T,128) bf16 rm; p2hat:(E,128) bf16 rm; hvT:(128,E) bf16 rm
__global__ __launch_bounds__(32)
void k_attn_out(const __bf16* __restrict__ P3, const __bf16* __restrict__ p2hat,
                const __bf16* __restrict__ hvT, const float* __restrict__ Hinc,
                const float* __restrict__ hold, const float* __restrict__ g,
                const float* __restrict__ bb, float* __restrict__ hnew) {
  const int lane = threadIdx.x;
  const int half = lane >> 4, ln16 = lane & 15;
  const int t0 = blockIdx.x * 16;

  __shared__ __attribute__((aligned(16))) __bf16 pl[16 * 32];

  // preload A fragments of this 16-row tile of P3 (full K=128)
  v16bf a3[4];
  {
    const __bf16* rowp = P3 + (size_t)(t0 + ln16) * Hh + half * 16;
#pragma unroll
    for (int k = 0; k < 4; ++k) a3[k] = load_a_frag(rowp + k * 32);
  }

  v8f O[8];
#pragma unroll
  for (int j = 0; j < 8; ++j) O[j] = zero8();
  float mrow[8], lrow[8];
#pragma unroll
  for (int r = 0; r < 8; ++r) { mrow[r] = -INFINITY; lrow[r] = 0.f; }

  for (int e0 = 0; e0 < Ee; e0 += 32) {
    // ---- scores: two 16x16 tiles (cols e0..e0+15 and e0+16..e0+31)
    v8f s0 = zero8(), s1 = zero8();
#pragma unroll
    for (int k = 0; k < 4; ++k) {
      v16bf b0 = *(const v16bf*)(p2hat + (size_t)(e0 + ln16) * Hh + k * 32 + half * 16);
      v16bf b1 = *(const v16bf*)(p2hat + (size_t)(e0 + 16 + ln16) * Hh + k * 32 + half * 16);
      s0 = wmma_bf16(a3[k], b0, s0);
      s1 = wmma_bf16(a3[k], b1, s1);
    }
    // ---- mask + online softmax
    float hm0[8], hm1[8];
#pragma unroll
    for (int r = 0; r < 8; ++r) {
      int t = t0 + r + half * 8;
      hm0[r] = Hinc[(size_t)t * Ee + e0 + ln16];
      hm1[r] = Hinc[(size_t)t * Ee + e0 + 16 + ln16];
      s0[r] = s0[r] * SCALE + (hm0[r] == 0.f ? NEGV : 0.f);
      s1[r] = s1[r] * SCALE + (hm1[r] == 0.f ? NEGV : 0.f);
    }
#pragma unroll
    for (int r = 0; r < 8; ++r) {
      float mx = fmaxf(s0[r], s1[r]);
      mx = fmaxf(mx, __shfl_xor(mx, 1, 32));
      mx = fmaxf(mx, __shfl_xor(mx, 2, 32));
      mx = fmaxf(mx, __shfl_xor(mx, 4, 32));
      mx = fmaxf(mx, __shfl_xor(mx, 8, 32));
      float nm = fmaxf(mrow[r], mx);
      float corr = __expf(mrow[r] - nm);
      float p0 = __expf(s0[r] - nm);
      float p1 = __expf(s1[r] - nm);
      float rs = p0 + p1;
      rs += __shfl_xor(rs, 1, 32);
      rs += __shfl_xor(rs, 2, 32);
      rs += __shfl_xor(rs, 4, 32);
      rs += __shfl_xor(rs, 8, 32);
      lrow[r] = lrow[r] * corr + rs;
      mrow[r] = nm;
#pragma unroll
      for (int j = 0; j < 8; ++j) O[j][r] *= corr;
      // beta = p * H_inc  ->  LDS as bf16 (C layout -> row-major 16x32 tile)
      pl[(r + half * 8) * 32 + ln16] = (__bf16)(p0 * hm0[r]);
      pl[(r + half * 8) * 32 + 16 + ln16] = (__bf16)(p1 * hm1[r]);
    }
    __syncthreads();
    // ---- reload beta tile as A-fragment (16x32)
    v16bf pa = load_a_frag(&pl[ln16 * 32 + half * 16]);
    __syncthreads();
    // ---- O += beta @ hv   (8 h-tiles, K = 32 edges)
#pragma unroll
    for (int j = 0; j < 8; ++j) {
      v16bf bh = *(const v16bf*)(hvT + (size_t)(j * 16 + ln16) * Ee + e0 + half * 16);
      O[j] = wmma_bf16(pa, bh, O[j]);
    }
  }

  // ---- epilogue: /l, +residual, layernorm
#pragma unroll
  for (int r = 0; r < 8; ++r) {
    float inv = 1.f / lrow[r];
#pragma unroll
    for (int j = 0; j < 8; ++j) O[j][r] *= inv;
  }
#pragma unroll
  for (int r = 0; r < 8; ++r) {
    int t = t0 + r + half * 8;
    float sum = 0.f;
#pragma unroll
    for (int j = 0; j < 8; ++j) {
      O[j][r] += hold[(size_t)t * Hh + j * 16 + ln16];
      sum += O[j][r];
    }
    sum += __shfl_xor(sum, 1, 32);
    sum += __shfl_xor(sum, 2, 32);
    sum += __shfl_xor(sum, 4, 32);
    sum += __shfl_xor(sum, 8, 32);
    float mu = sum * (1.f / (float)Hh);
    float vs = 0.f;
#pragma unroll
    for (int j = 0; j < 8; ++j) { float d = O[j][r] - mu; vs += d * d; }
    vs += __shfl_xor(vs, 1, 32);
    vs += __shfl_xor(vs, 2, 32);
    vs += __shfl_xor(vs, 4, 32);
    vs += __shfl_xor(vs, 8, 32);
    float rstd = rsqrtf(vs * (1.f / (float)Hh) + EPSV);
#pragma unroll
    for (int j = 0; j < 8; ++j) {
      int h = j * 16 + ln16;
      hnew[(size_t)t * Hh + h] = (O[j][r] - mu) * rstd * g[h] + bb[h];
    }
  }
}

// ---------------- host orchestration ---------------------------------------
extern "C" void kernel_launch(void* const* d_in, const int* in_sizes, int n_in,
                              void* d_out, int out_size, void* d_ws, size_t ws_size,
                              hipStream_t stream) {
  (void)in_sizes; (void)n_in; (void)out_size; (void)ws_size;
  const float* node = (const float*)d_in[0];
  const float* Hinc = (const float*)d_in[1];
  const int*   memb = (const int*)d_in[2];
  const float* W1   = (const float*)d_in[3];
  const float* W1h  = (const float*)d_in[4];
  const float* ctx1 = (const float*)d_in[5];
  const float* W2   = (const float*)d_in[6];
  const float* W2h  = (const float*)d_in[7];
  const float* W3   = (const float*)d_in[8];
  const float* lng  = (const float*)d_in[9];
  const float* lnb  = (const float*)d_in[10];
  const float* rini = (const float*)d_in[11];
  float* out = (float*)d_out;

  char* wptr = (char*)d_ws;
  auto take = [&](size_t bytes) -> void* {
    void* p = (void*)wptr;
    wptr += (bytes + 255) & ~(size_t)255;
    return p;
  };
  float*  hA0    = (float*)take((size_t)Tt * Hh * 4);
  float*  hA1    = (float*)take((size_t)Tt * Hh * 4);
  __bf16* hAbf   = (__bf16*)take((size_t)Tt * Hh * 2);
  __bf16* w1bf   = (__bf16*)take((size_t)Hh * Hh * 2);
  __bf16* w3bf   = (__bf16*)take((size_t)Hh * Hh * 2);
  __bf16* w2hbf  = (__bf16*)take((size_t)Hh * Hh * 2);
  __bf16* w2bf   = (__bf16*)take((size_t)Hh * Hh * 2);
  float*  c1     = (float*)take((size_t)Hh * 4);
  float*  u      = (float*)take((size_t)Tt * 4);
  float*  sv     = (float*)take((size_t)Ee * 4);
  __bf16* AT     = (__bf16*)take((size_t)Ee * Tp * 2);
  __bf16* P1T    = (__bf16*)take((size_t)Hh * Tp * 2);
  __bf16* P3     = (__bf16*)take((size_t)Tt * Hh * 2);
  __bf16* hypbf  = (__bf16*)take((size_t)Ee * Hh * 2);
  __bf16* p2hat  = (__bf16*)take((size_t)Ee * Hh * 2);
  __bf16* hvT    = (__bf16*)take((size_t)Hh * Ee * 2);

  const int cvtTH = (Tt * Hh + 255) / 256;
  const int cvtHH = (Hh * Hh + 255) / 256;

  for (int b = 0; b < Bc; ++b) {
    float* hcur = hA0;
    float* hnext = hA1;
    k_region<<<Rr, 128, 0, stream>>>(node + (size_t)b * Nn * Hh, memb, rini, hcur);
    k_copy_f32<<<512, 256, 0, stream>>>(node + (size_t)b * Nn * Hh, hcur, Nn * Hh);
    for (int l = 0; l < Ll; ++l) {
      k_cvt_bf16<<<cvtTH, 256, 0, stream>>>(hcur, hAbf, Tt * Hh);
      k_cvt_bf16<<<cvtHH, 256, 0, stream>>>(W1 + (size_t)l * Hh * Hh, w1bf, Hh * Hh);
      k_cvt_bf16<<<cvtHH, 256, 0, stream>>>(W3 + (size_t)l * Hh * Hh, w3bf, Hh * Hh);
      k_cvt_bf16<<<cvtHH, 256, 0, stream>>>(W2h + (size_t)l * Hh * Hh, w2hbf, Hh * Hh);
      k_cvt_bf16<<<cvtHH, 256, 0, stream>>>(W2 + (size_t)l * Hh * Hh, w2bf, Hh * Hh);
      k_c1<<<1, 128, 0, stream>>>(W1h + (size_t)l * Hh * Hh, ctx1 + (size_t)l * Hh, c1);
      k_raw_u<<<1, 1024, 0, stream>>>(hcur, c1, u);
      k_AbfT<<<dim3(Tp / 32, Ee / 32), dim3(32, 8), 0, stream>>>(Hinc, u, AT);
      k_denom<<<Ee / 256, 256, 0, stream>>>(Hinc, u, sv);
      // P1T = (h_aug @ W1^T)^T  (H x Tp, bf16), zero K-pad
      k_gemm_xwT<<<Tt / 16, 256, 0, stream>>>(hAbf, w1bf, nullptr, P1T, Tp);
      k_zero_tail<<<(Hh * (Tp - Tt) + 255) / 256, 256, 0, stream>>>(P1T, Tp, Hh, Tt, Tp);
      // P3 = h_aug @ W3^T  (T x H, bf16)
      k_gemm_xwT<<<Tt / 16, 256, 0, stream>>>(hAbf, w3bf, P3, nullptr, 0);
      // hyper (E x H, bf16, row-scaled by s) -- TDM-staged A tiles
      k_hyper<<<Ee / 16, 256, 0, stream>>>(AT, P1T, sv, hypbf);
      // proj2_hat = hyper @ W2_hat^T ; hvT = (hyper @ W2^T)^T
      k_gemm_xwT<<<Ee / 16, 256, 0, stream>>>(hypbf, w2hbf, p2hat, nullptr, 0);
      k_gemm_xwT<<<Ee / 16, 256, 0, stream>>>(hypbf, w2bf, nullptr, hvT, Ee);
      // fused stage2 + residual + layernorm
      k_attn_out<<<Tt / 16, 32, 0, stream>>>(P3, p2hat, hvT, Hinc, hcur,
                                             lng + (size_t)l * Hh, lnb + (size_t)l * Hh,
                                             hnext);
      float* tmp = hcur; hcur = hnext; hnext = tmp;
    }
    k_copy_f32<<<512, 256, 0, stream>>>(hcur, out + (size_t)b * Nn * Hh, Nn * Hh);
  }
}
